// JKNet_91207925498527
// MI455X (gfx1250) — matrix operand order repaired and checked by reading, not compiled, not run on hardware
//
#include <hip/hip_runtime.h>
#include <math.h>

typedef __attribute__((ext_vector_type(16))) _Float16 v16h;
typedef __attribute__((ext_vector_type(8)))  _Float16 v8h;
typedef __attribute__((ext_vector_type(8)))  float    v8f;
typedef _Float16 h16;

#define DIN  256
#define HID  128
#define NCLS 41
#define NPAD 48

// ---------------- small utility kernels ----------------

__global__ void k_fill1(float* __restrict__ p, int n) {
  int i = blockIdx.x * blockDim.x + threadIdx.x;
  if (i < n) p[i] = 1.0f;   // self-loop contributes 1 to every degree
}

__global__ void k_degree(const int* __restrict__ dst, float* __restrict__ deg, int E) {
  int i = blockIdx.x * blockDim.x + threadIdx.x;
  if (i < E) atomicAdd(&deg[dst[i]], 1.0f);
}

__global__ void k_rsqrt(float* __restrict__ p, int n) {
  int i = blockIdx.x * blockDim.x + threadIdx.x;
  if (i < n) p[i] = rsqrtf(p[i]);   // deg >= 1 always (self loops)
}

__global__ void k_init_bias(float* __restrict__ buf, const float* __restrict__ b, int n) {
  for (int i = blockIdx.x * blockDim.x + threadIdx.x; i < n; i += gridDim.x * blockDim.x)
    buf[i] = b[i & (HID - 1)];
}

__global__ void k_relu(float* __restrict__ p, int n) {
  for (int i = blockIdx.x * blockDim.x + threadIdx.x; i < n; i += gridDim.x * blockDim.x)
    p[i] = fmaxf(p[i], 0.0f);
}

// W [K, Nout] f32 row-major  ->  Wt [Npad, K] f16 (transposed, zero-padded cols)
__global__ void k_convert_wt(const float* __restrict__ W, h16* __restrict__ Wt,
                             int K, int Nout, int Npad) {
  int i = blockIdx.x * blockDim.x + threadIdx.x;
  if (i >= K * Npad) return;
  int n = i / K, k = i - n * K;
  Wt[i] = (n < Nout) ? (h16)W[k * Nout + n] : (h16)0.0f;
}

// ---------------- WMMA GEMM: C[M, ldc] = A[M,K] (+A2) @ Bt^T ----------------
// Bt is [NT*16, K] f16 (transposed weights). One wave per 16-col tile.
// All indexing is 32-bit (M*K, M*ldc < 2^31 here).
template<int K, int NT, bool ASUM>
__global__ __launch_bounds__(32 * NT)
void k_gemm(const float* __restrict__ A, const float* __restrict__ A2,
            const h16* __restrict__ Bt, float* __restrict__ C,
            int M, int ldc) {
  constexpr int LK = K + 8;                       // LDS row pad: breaks bank conflicts
  __shared__ alignas(16) h16 lds_b[NT * 16 * LK];
  __shared__ alignas(16) h16 lds_a[16 * LK];

  const int tid  = threadIdx.x;
  const int lane = tid & 31;
  const int wave = tid >> 5;                      // N-tile index
  const int m0   = blockIdx.x * 16;
  const bool full = (m0 + 16 <= M);               // block-uniform (true for M%16==0)

  // stage full Bt into LDS (L2-resident weights)
  #pragma unroll 4
  for (int i = tid; i < NT * 16 * K; i += 32 * NT) {
    int n = i / K, k = i - n * K;
    lds_b[n * LK + k] = Bt[i];
  }
  // stage A panel (16 rows x K), f32 -> f16, optional fused x1+x2
  if (full) {
    const float* Ab  = A + m0 * K;
    const float* A2b = ASUM ? A2 + m0 * K : nullptr;
    #pragma unroll 4
    for (int i = tid; i < 16 * K; i += 32 * NT) {
      int r = i / K, c = i - r * K;
      float v = Ab[i];
      if (ASUM) v += A2b[i];
      lds_a[r * LK + c] = (h16)v;
    }
  } else {
    for (int i = tid; i < 16 * K; i += 32 * NT) {
      int r = i / K, c = i - r * K;
      int row = m0 + r;
      float v = 0.0f;
      if (row < M) {
        v = A[row * K + c];
        if (ASUM) v += A2[row * K + c];
      }
      lds_a[r * LK + c] = (h16)v;
    }
  }
  if (full && m0 + 16 < M) __builtin_prefetch(&A[(m0 + 16) * K + lane], 0, 0);
  __syncthreads();

  const int r  = lane & 15;                       // A row / B col / D col
  const int hi = lane >> 4;                       // lane-half selector
  v8f acc = {};

  #pragma unroll
  for (int k0 = 0; k0 < K; k0 += 32) {
    // A 16x32 f16 frag: lanes<16 hold K {0-7,16-23}; lanes>=16 hold {8-15,24-31}
    v8h alo = *(const v8h*)&lds_a[r * LK + k0 + hi * 8];
    v8h ahi = *(const v8h*)&lds_a[r * LK + k0 + 16 + hi * 8];
    // B 32x16 f16 frag: N = lane&15, K same as A: {8*hi + j, 16 + 8*hi + j}  (Bt stored [n][k])
    v8h blo = *(const v8h*)&lds_b[(wave * 16 + r) * LK + k0 + hi * 8];
    v8h bhi = *(const v8h*)&lds_b[(wave * 16 + r) * LK + k0 + 16 + hi * 8];
    v16h af, bf;
    #pragma unroll
    for (int j = 0; j < 8; ++j) {
      af[j] = alo[j]; af[8 + j] = ahi[j];
      bf[j] = blo[j]; bf[8 + j] = bhi[j];
    }
    acc = __builtin_amdgcn_wmma_f32_16x16x32_f16(false, af, false, bf,
                                                 (short)0, acc, false, false);
  }

  // D layout: VGPR v -> M = 8*hi + v, N = lane&15
  const int n0 = wave * 16 + r;
  float* cp = C + (m0 + hi * 8) * ldc + n0;       // 32-bit base, constant row offsets
  if (full) {
    #pragma unroll
    for (int v = 0; v < 8; ++v) cp[v * ldc] = acc[v];
  } else {
    #pragma unroll
    for (int v = 0; v < 8; ++v)
      if (m0 + hi * 8 + v < M) cp[v * ldc] = acc[v];
  }
}

// ---------------- edge gather-scale-scatter (one wave per edge) ----------------
__global__ void k_scatter(const int* __restrict__ src, const int* __restrict__ dst,
                          const float* __restrict__ dis, const float* __restrict__ h,
                          float* __restrict__ out, int E, int Nn) {
  const int lane = threadIdx.x & 31;
  const int wpb  = blockDim.x >> 5;
  int w        = blockIdx.x * wpb + (threadIdx.x >> 5);
  const int nw = gridDim.x * wpb;
  const int total = E + Nn;                       // real edges + self loops
  for (int e = w; e < total; e += nw) {
    int s, d;
    if (e < E) { s = src[e]; d = dst[e]; } else { s = e - E; d = s; }
    const float nrm = dis[s] * dis[d];
    const float4 v = ((const float4*)(h + (size_t)s * HID))[lane];
    float* o = out + (size_t)d * HID + lane * 4;
    atomicAdd(o + 0, v.x * nrm);
    atomicAdd(o + 1, v.y * nrm);
    atomicAdd(o + 2, v.z * nrm);
    atomicAdd(o + 3, v.w * nrm);
  }
}

// ---------------- per-row log_softmax over 41 classes (one wave per row) ----------------
__global__ void k_logsoftmax(const float* __restrict__ z, const float* __restrict__ bias,
                             float* __restrict__ out, int M) {
  const int lane = threadIdx.x & 31;
  const int wpb  = blockDim.x >> 5;
  int w        = blockIdx.x * wpb + (threadIdx.x >> 5);
  const int nw = gridDim.x * wpb;
  for (int row = w; row < M; row += nw) {
    const float* zr = z + (size_t)row * NPAD;
    float v0 = (lane < NCLS)      ? zr[lane]      + bias[lane]      : -1e30f;
    float v1 = (lane + 32 < NCLS) ? zr[lane + 32] + bias[lane + 32] : -1e30f;
    float m = fmaxf(v0, v1);
    #pragma unroll
    for (int o = 16; o > 0; o >>= 1) m = fmaxf(m, __shfl_xor(m, o, 32));
    float s = 0.0f;
    if (lane < NCLS)      s += __expf(v0 - m);
    if (lane + 32 < NCLS) s += __expf(v1 - m);
    #pragma unroll
    for (int o = 16; o > 0; o >>= 1) s += __shfl_xor(s, o, 32);
    const float ls = __logf(s);
    float* orow = out + (size_t)row * NCLS;
    if (lane < NCLS)      orow[lane]      = v0 - m - ls;
    if (lane + 32 < NCLS) orow[lane + 32] = v1 - m - ls;
  }
}

// ---------------- driver ----------------
extern "C" void kernel_launch(void* const* d_in, const int* in_sizes, int n_in,
                              void* d_out, int out_size, void* d_ws, size_t ws_size,
                              hipStream_t stream) {
  const float* x  = (const float*)d_in[0];
  const int*   ei = (const int*)  d_in[1];   // [2, E] int
  const float* W1 = (const float*)d_in[2];
  const float* b1 = (const float*)d_in[3];
  const float* W2 = (const float*)d_in[4];
  const float* b2 = (const float*)d_in[5];
  const float* lw = (const float*)d_in[6];
  const float* lb = (const float*)d_in[7];

  const int N = in_sizes[0] / DIN;
  const int E = in_sizes[1] / 2;
  const int* src = ei;
  const int* dst = ei + E;

  auto align256 = [](size_t v) { return (v + 255) & ~(size_t)255; };
  char* w = (char*)d_ws;
  float* deg  = (float*)w;  w += align256((size_t)N * 4);              // becomes dis (in-place)
  h16*   W1t  = (h16*)w;    w += align256((size_t)HID * DIN * 2);
  h16*   W2t  = (h16*)w;    w += align256((size_t)HID * HID * 2);
  h16*   LWt  = (h16*)w;    w += align256((size_t)NPAD * HID * 2);
  float* hbuf = (float*)w;  w += align256((size_t)N * HID * 4);
  float* x1   = (float*)w;  w += align256((size_t)N * HID * 4);
  float* x2   = (float*)w;  w += align256((size_t)N * HID * 4);
  float* z    = (float*)w;  w += align256((size_t)N * NPAD * 4);
  (void)ws_size; (void)n_in; (void)out_size;

  // symmetric normalization
  k_fill1 <<<(N + 255) / 256, 256, 0, stream>>>(deg, N);
  k_degree<<<(E + 255) / 256, 256, 0, stream>>>(dst, deg, E);
  k_rsqrt <<<(N + 255) / 256, 256, 0, stream>>>(deg, N);

  // pre-convert weights to transposed f16
  k_convert_wt<<<(DIN * HID + 255) / 256, 256, 0, stream>>>(W1, W1t, DIN, HID, HID);
  k_convert_wt<<<(HID * HID + 255) / 256, 256, 0, stream>>>(W2, W2t, HID, HID, HID);
  k_convert_wt<<<(HID * NPAD + 255) / 256, 256, 0, stream>>>(lw, LWt, HID, NCLS, NPAD);

  const int mb = (N + 15) / 16;

  // conv1: h = x @ W1 ; x1 = relu(scatter(h) + b1)
  k_gemm<DIN, HID / 16, false><<<mb, 32 * (HID / 16), 0, stream>>>(x, nullptr, W1t, hbuf, N, HID);
  k_init_bias<<<8192, 256, 0, stream>>>(x1, b1, N * HID);
  k_scatter<<<4096, 256, 0, stream>>>(src, dst, deg, hbuf, x1, E, N);
  k_relu<<<8192, 256, 0, stream>>>(x1, N * HID);

  // conv2: h = x1 @ W2 ; x2 = relu(scatter(h) + b2)
  k_gemm<HID, HID / 16, false><<<mb, 32 * (HID / 16), 0, stream>>>(x1, nullptr, W2t, hbuf, N, HID);
  k_init_bias<<<8192, 256, 0, stream>>>(x2, b2, N * HID);
  k_scatter<<<4096, 256, 0, stream>>>(src, dst, deg, hbuf, x2, E, N);
  k_relu<<<8192, 256, 0, stream>>>(x2, N * HID);

  // head: z = (x1 + x2) @ lin_w (padded to 48 cols) ; out = log_softmax(z + lin_b)
  k_gemm<HID, NPAD / 16, true><<<mb, 32 * (NPAD / 16), 0, stream>>>(x1, x2, LWt, z, N, NPAD);
  k_logsoftmax<<<2048, 256, 0, stream>>>(z, lb, (float*)d_out, N);
}
